// Qwen3_5GatedDeltaNet_22196390986345
// MI455X (gfx1250) — matrix operand order repaired and checked by reading, not compiled, not run on hardware
//
#include <hip/hip_runtime.h>
#include <hip/hip_bf16.h>

typedef unsigned short u16;
typedef unsigned int   u32;
typedef __attribute__((ext_vector_type(16))) __bf16 v16bf;
typedef __attribute__((ext_vector_type(8)))  float  v8f;
typedef __attribute__((ext_vector_type(8)))  u32    v8u;
typedef __attribute__((ext_vector_type(4)))  u32    v4u;
typedef __attribute__((ext_vector_type(8)))  int    v8i;
typedef __attribute__((ext_vector_type(4)))  int    v4i;

#define H_DIM     2048
#define HK        16
#define HV        32
#define DK        128
#define DV        128
#define KEY_DIM   2048
#define VAL_DIM   4096
#define CONV_DIM  8192
#define QKVZ_DIM  12288
#define T_LEN     4096
#define CHUNK     16
#define NCHUNK    (T_LEN / CHUNK)

// ---- CDNA5 feature switches --------------------------------------------------
#define GEMM_ASYNC 1   // global_load_async_to_lds_b128 staging in GEMM (asm-verified)

#if defined(__has_builtin)
#if __has_builtin(__builtin_amdgcn_tensor_load_to_lds)
#define HAVE_TDM 1
#endif
#endif
#ifndef HAVE_TDM
#define HAVE_TDM 0
#endif

// Compile-time diagnostics (stderr is our only telemetry).
#define CDNA5_STR2(x) #x
#define CDNA5_STR(x) CDNA5_STR2(x)
#pragma message("CDNA5 INFO: clang major = " CDNA5_STR(__clang_major__))
#if HAVE_TDM
#pragma message("CDNA5 INFO: TDM tensor_load_to_lds builtin ACTIVE")
#else
#warning "CDNA5 INFO: TDM builtin NOT available; recurrence uses fallback vector loads"
#endif

__device__ __forceinline__ u16 f2b(float f) {
  u32 u = __float_as_uint(f);
  u32 r = u + 0x7FFFu + ((u >> 16) & 1u);
  return (u16)(r >> 16);
}
__device__ __forceinline__ float b2f(u16 h) { return __uint_as_float(((u32)h) << 16); }

__device__ __forceinline__ v8f vzero8() {
  v8f r;
#pragma unroll
  for (int i = 0; i < 8; ++i) r[i] = 0.f;
  return r;
}

__device__ __forceinline__ v8f wmma_bf16(v16bf a, v16bf b, v8f c) {
  return __builtin_amdgcn_wmma_f32_16x16x32_bf16(false, a, false, b, (short)0, c, false, false);
}

// Low 32 bits of a generic pointer to LDS == LDS byte offset (aperture form).
__device__ __forceinline__ u32 lds_addr(const void* p) { return (u32)(size_t)p; }

template <int N>
__device__ __forceinline__ void wait_async() {
  asm volatile("s_wait_asynccnt %0" ::"i"(N) : "memory");
}
__device__ __forceinline__ void async_b128(u32 ldsoff, const void* g) {
  asm volatile("global_load_async_to_lds_b128 %0, %1, off" ::"v"(ldsoff), "v"(g) : "memory");
}

#if HAVE_TDM
// TDM: load a 16-row x 128-halfword bf16 tile (row stride in elements) into LDS.
__device__ __forceinline__ void tdm_load_tile(u32 lds_off, const u16* g, u32 stride_elems) {
  unsigned long long ga = (unsigned long long)(size_t)g;
  v4u g0;
  g0[0] = 1u;                                // count=1, user descriptor
  g0[1] = lds_off;                           // lds_addr
  g0[2] = (u32)ga;                           // global_addr[31:0]
  g0[3] = (u32)(ga >> 32) | 0x80000000u;     // global_addr[56:32] | type=2("image")
  v8i g1;
  g1[0] = 0x00010000;                        // workgroup_mask=0, data_size=1 (2B)
  g1[1] = (int)(128u << 16);                 // tensor_dim0 = 128
  g1[2] = (int)(16u << 16);                  // tensor_dim1 = 16
  g1[3] = (int)(128u << 16);                 // tile_dim0 = 128
  g1[4] = 16;                                // tile_dim1 = 16
  g1[5] = (int)stride_elems;                 // tensor_dim0_stride[31:0]
  g1[6] = 0;
  g1[7] = 0;
  v4i z4; z4[0] = 0; z4[1] = 0; z4[2] = 0; z4[3] = 0;
#if __clang_major__ >= 23
  v8i z8; z8[0]=0; z8[1]=0; z8[2]=0; z8[3]=0; z8[4]=0; z8[5]=0; z8[6]=0; z8[7]=0;
  __builtin_amdgcn_tensor_load_to_lds(g0, g1, z4, z4, z8, 0);
#else
  __builtin_amdgcn_tensor_load_to_lds(g0, g1, z4, z4, 0);
#endif
}
__device__ __forceinline__ void wait_tensor0() {
#if __has_builtin(__builtin_amdgcn_s_wait_tensorcnt)
  __builtin_amdgcn_s_wait_tensorcnt(0);
#else
  asm volatile("s_wait_tensorcnt 0x0" ::: "memory");
#endif
}
#endif

// A fragment: 16x32 bf16 tile, row-major [M][K] in LDS (ld halfwords, even).
__device__ __forceinline__ v16bf lda_frag(const u16* base, int ld) {
  int lane = threadIdx.x & 31;
  int row  = lane & 15;
  int kb2  = (lane >> 4) << 2;
  const u32* p = (const u32*)(base + row * ld);
  v8u w;
#pragma unroll
  for (int j = 0; j < 4; ++j) w[j] = p[kb2 + j];
#pragma unroll
  for (int j = 0; j < 4; ++j) w[4 + j] = p[8 + kb2 + j];
  return __builtin_bit_cast(v16bf, w);
}
// B fragment: 32x16 tile stored TRANSPOSED [N][K] in LDS (ld halfwords).
__device__ __forceinline__ v16bf ldb_frag(const u16* base, int ld) {
  int lane = threadIdx.x & 31;
  int n    = lane & 15;
  int kb   = (lane >> 4) << 3;
  const u32* p = (const u32*)(base + n * ld + kb);
  v8u w;
#pragma unroll
  for (int j = 0; j < 4; ++j) w[j] = p[j];
#pragma unroll
  for (int j = 0; j < 4; ++j) w[4 + j] = p[8 + j];
  return __builtin_bit_cast(v16bf, w);
}

// ---------------------------------------------------------------- f32 -> bf16
__global__ __launch_bounds__(256) void cvt_bf16_kernel(const float* __restrict__ s,
                                                       u16* __restrict__ d, long n) {
  long i = (long)blockIdx.x * 256 + threadIdx.x;
  if (i < n) d[i] = f2b(s[i]);
}

// ------------------------------------------------- f32 [K][N] -> bf16 [N][K] transpose
__global__ __launch_bounds__(256) void cvt_t_kernel(const float* __restrict__ src,
                                                    u16* __restrict__ dst, int K, int N) {
  __shared__ float tile[32][33];
  int nb = blockIdx.x * 32, kb = blockIdx.y * 32;
  int tx = threadIdx.x & 31, ty = threadIdx.x >> 5;
#pragma unroll
  for (int i = 0; i < 32; i += 8)
    tile[ty + i][tx] = src[(size_t)(kb + ty + i) * N + nb + tx];
  __syncthreads();
#pragma unroll
  for (int i = 0; i < 32; i += 8)
    dst[(size_t)(nb + ty + i) * K + kb + tx] = f2b(tile[tx][ty + i]);
}

// ---------------------------------------------------------------- bf16 WMMA GEMM
// C[M,N] = A[M,K] @ Bt[N,K]^T ; A,Bt bf16, C f32. M%128==0, K%32==0, N guarded.
// Double-buffered LDS staging via async-to-LDS copies (4 b128 per thread per tile).
__global__ __launch_bounds__(256) void gemm_bf16_kernel(const u16* __restrict__ A,
                                                        const u16* __restrict__ Bt,
                                                        float* __restrict__ C,
                                                        int M, int N, int K) {
  const int LDA = 40;
  __shared__ u16 As[2][128 * 40];
  __shared__ u16 Bs[2][128 * 40];
  int tid  = threadIdx.x;
  int lane = tid & 31;
  int wv   = tid >> 5;
  int wm   = wv >> 1;
  int wn   = wv & 1;
  int m0   = blockIdx.y * 128;
  int n0   = blockIdx.x * 128;

  int r  = tid >> 1;
  int q0 = (tid & 1) << 1;  // this thread stages quarters q0, q0+1 of its row
  const u16* arow = A + (size_t)(m0 + r) * K;
  int bi = n0 + r; if (bi > N - 1) bi = N - 1;  // clamp OOB rows (cols guarded at store)
  const u16* brow = Bt + (size_t)bi * K;

  v8f acc[2][4];
  for (int mf = 0; mf < 2; ++mf)
    for (int nf = 0; nf < 4; ++nf) acc[mf][nf] = vzero8();

  auto stage = [&](int buf, int k0) {
#pragma unroll
    for (int j = 0; j < 2; ++j) {
      int q = q0 + j;
#if GEMM_ASYNC
      async_b128(lds_addr(&As[buf][r * LDA + q * 8]), arow + k0 + q * 8);
      async_b128(lds_addr(&Bs[buf][r * LDA + q * 8]), brow + k0 + q * 8);
#else
      *(uint4*)&As[buf][r * LDA + q * 8] = *(const uint4*)(arow + k0 + q * 8);
      *(uint4*)&Bs[buf][r * LDA + q * 8] = *(const uint4*)(brow + k0 + q * 8);
#endif
    }
  };

  int nk = K >> 5;
  stage(0, 0);
  for (int t = 0; t < nk; ++t) {
    int buf = t & 1;
    if (t + 1 < nk) {
      stage(buf ^ 1, (t + 1) << 5);
#if GEMM_ASYNC
      wait_async<4>();  // 8 outstanding, oldest 4 (this buf) retired
#endif
    } else {
#if GEMM_ASYNC
      wait_async<0>();
#endif
    }
    __syncthreads();
    v16bf af[2];
    for (int mf = 0; mf < 2; ++mf)
      af[mf] = lda_frag(&As[buf][(wm * 32 + mf * 16) * LDA], LDA);
    for (int nf = 0; nf < 4; ++nf) {
      v16bf bfr = ldb_frag(&Bs[buf][(wn * 64 + nf * 16) * LDA], LDA);
      for (int mf = 0; mf < 2; ++mf) acc[mf][nf] = wmma_bf16(af[mf], bfr, acc[mf][nf]);
    }
    __syncthreads();
  }
  int hi = (lane >> 4) << 3;
  int cl = lane & 15;
  for (int mf = 0; mf < 2; ++mf)
    for (int nf = 0; nf < 4; ++nf) {
      int col = n0 + wn * 64 + nf * 16 + cl;
      if (col < N) {
        for (int i = 0; i < 8; ++i) {
          int row = m0 + wm * 32 + mf * 16 + hi + i;
          C[(size_t)row * N + col] = acc[mf][nf][i];
        }
      }
    }
}

// ---------------------------------------------------------------- conv + SiLU
// 4 timesteps per thread: 7 row-reads for 4 outputs (vs 4 reads per output).
__global__ __launch_bounds__(256) void conv_silu_kernel(const float* __restrict__ qkvz,
                                                        const float* __restrict__ conv_w,
                                                        float* __restrict__ qkf,
                                                        u16* __restrict__ Vb) {
  int idx = blockIdx.x * 256 + threadIdx.x;  // (T_LEN/4)*CONV_DIM threads
  int tq  = idx >> 13;                       // t-quad
  int c   = idx & 8191;
  int t0  = tq << 2;
  float w0 = conv_w[c * 4 + 0], w1 = conv_w[c * 4 + 1];
  float w2 = conv_w[c * 4 + 2], w3 = conv_w[c * 4 + 3];
  float x[7];
#pragma unroll
  for (int j = 0; j < 7; ++j) {
    int t = t0 - 3 + j;
    x[j] = (t >= 0) ? qkvz[(size_t)t * QKVZ_DIM + c] : 0.f;
  }
#pragma unroll
  for (int i = 0; i < 4; ++i) {
    float acc = w0 * x[i] + w1 * x[i + 1] + w2 * x[i + 2] + w3 * x[i + 3];
    float y = acc / (1.f + __expf(-acc));
    int t = t0 + i;
    if (c < 4096) qkf[(size_t)t * 4096 + c] = y;
    else          Vb[(size_t)t * 4096 + (c - 4096)] = f2b(y);
  }
}

// ---------------------------------------------------------------- q/k L2 norm
__global__ __launch_bounds__(128) void qknorm_kernel(const float* __restrict__ qkf,
                                                     u16* __restrict__ Qb,
                                                     u16* __restrict__ Kb) {
  int t  = blockIdx.x;
  int hh = blockIdx.y;
  int v  = threadIdx.x;
  int lane = v & 31, wv = v >> 5;
  float x  = qkf[(size_t)t * 4096 + hh * 128 + v];
  float ss = x * x;
#pragma unroll
  for (int off = 16; off > 0; off >>= 1) ss += __shfl_xor(ss, off, 32);
  __shared__ float red[4];
  if (lane == 0) red[wv] = ss;
  __syncthreads();
  float tot = red[0] + red[1] + red[2] + red[3];
  float sc  = rsqrtf(tot + 1e-6f);
  if (hh < 16)
    Qb[(size_t)t * KEY_DIM + hh * 128 + v] = f2b(x * sc * 0.08838834764831845f);
  else
    Kb[(size_t)t * KEY_DIM + (hh - 16) * 128 + v] = f2b(x * sc);
}

// ---------------------------------------------------------------- g / beta
__global__ __launch_bounds__(256) void gba_kernel(const float* __restrict__ ba,
                                                  const float* __restrict__ A_log,
                                                  const float* __restrict__ dt_bias,
                                                  float* __restrict__ gbuf,
                                                  float* __restrict__ betabuf) {
  int idx = blockIdx.x * 256 + threadIdx.x;
  if (idx >= T_LEN * HV) return;
  int t = idx >> 5, h = idx & 31;
  float b  = ba[(size_t)t * 64 + h];
  float a  = ba[(size_t)t * 64 + 32 + h];
  float x  = a + dt_bias[h];
  float sp = (x > 20.f) ? x : log1pf(__expf(x));
  gbuf[idx]    = -__expf(A_log[h]) * sp;
  betabuf[idx] = 1.f / (1.f + __expf(-b));
}

// ---------------------------------------------------------------- chunked gated delta rule
// One block per value head; 128 threads (4 waves). Chunk C=16.
// Q/K/V chunk tiles are DMA'd (TDM when available) one chunk ahead, ping-pong buffers.
// bf16 state copy (Sb) is maintained incrementally by the state-update stage.
__global__ __launch_bounds__(128) void recurrence_kernel(const u16* __restrict__ Qb,
                                                         const u16* __restrict__ Kb,
                                                         const u16* __restrict__ Vb,
                                                         const float* __restrict__ gbuf,
                                                         const float* __restrict__ betabuf,
                                                         float* __restrict__ o) {
  extern __shared__ char smem[];
  float* Sf   = (float*)smem;                        // [128][128] state, Sf[v*128+d]
  u16*   Sb   = (u16*)(smem + 65536);                // bf16 copy [v][d]
  u16*   TQ0  = (u16*)(smem + 98304);                // ping-pong raw chunk tiles [16][128]
  u16*   TK0  = TQ0 + 2048;
  u16*   TV0  = TK0 + 2048;
  u16*   TQ1  = TV0 + 2048;
  u16*   TK1  = TQ1 + 2048;
  u16*   TV1  = TK1 + 2048;
  u16*   Qs   = TV1 + 2048;                          // lambda-scaled q [16][128]
  u16*   Ks   = Qs + 2048;                           // lambda-scaled k
  float* RHS  = (float*)(Ks + 2048);                 // (L*K)@S0 [16][128]
  float* Oin  = RHS + 2048;                          // (L*Q)@S0
  float* Pqk  = Oin + 2048;                          // QK^T [16][16]
  float* Pkk  = Pqk + 256;
  float* Msub = Pkk + 256;
  u16*   Atl  = (u16*)(Msub + 256);                  // tril(D*QK^T) [16][32] K-padded
  u16*   Ubt  = Atl + 512;                           // U^T [v][s] [128][32] K-padded
  u16*   Kh   = Ubt + 4096;                          // (lamC/lam_s)k_s[d] [128][32]
  float* Gc   = (float*)(Kh + 4096);
  float* lam  = Gc + 16;
  float* rdec = lam + 16;
  float* bet  = rdec + 16;

  int h   = blockIdx.x;
  int kh  = h >> 1;
  int tid = threadIdx.x;
  int lane = tid & 31;
  int wv   = tid >> 5;

  for (int i = tid; i < 128 * 128; i += 128) { Sf[i] = 0.f; Sb[i] = 0; }
  for (int i = tid; i < 128 * 32; i += 128) { Ubt[i] = 0; Kh[i] = 0; }
  for (int i = tid; i < 16 * 32; i += 128) Atl[i] = 0;
  __syncthreads();

  auto issue = [&](int t0, u16* tq, u16* tk, u16* tv) {
    const u16* qg = Qb + (size_t)t0 * KEY_DIM + kh * DK;
    const u16* kg = Kb + (size_t)t0 * KEY_DIM + kh * DK;
    const u16* vg = Vb + (size_t)t0 * VAL_DIM + h * DV;
#if HAVE_TDM
    if (wv == 0) {
      tdm_load_tile(lds_addr(tq), qg, KEY_DIM);
      tdm_load_tile(lds_addr(tk), kg, KEY_DIM);
      tdm_load_tile(lds_addr(tv), vg, VAL_DIM);
    }
#else
    int r = tid >> 3, cs = (tid & 7) << 4;
    *(uint4*)(tq + r * 128 + cs)     = *(const uint4*)(qg + (size_t)r * KEY_DIM + cs);
    *(uint4*)(tq + r * 128 + cs + 8) = *(const uint4*)(qg + (size_t)r * KEY_DIM + cs + 8);
    *(uint4*)(tk + r * 128 + cs)     = *(const uint4*)(kg + (size_t)r * KEY_DIM + cs);
    *(uint4*)(tk + r * 128 + cs + 8) = *(const uint4*)(kg + (size_t)r * KEY_DIM + cs + 8);
    *(uint4*)(tv + r * 128 + cs)     = *(const uint4*)(vg + (size_t)r * VAL_DIM + cs);
    *(uint4*)(tv + r * 128 + cs + 8) = *(const uint4*)(vg + (size_t)r * VAL_DIM + cs + 8);
#endif
  };

  issue(0, TQ0, TK0, TV0);

  for (int ch = 0; ch < NCHUNK; ++ch) {
    int t0 = ch * CHUNK;
    u16* tq = (ch & 1) ? TQ1 : TQ0;
    u16* tk = (ch & 1) ? TK1 : TK0;
    u16* tv = (ch & 1) ? TV1 : TV0;
#if HAVE_TDM
    if (wv == 0) wait_tensor0();
#endif
    __syncthreads();  // current chunk tiles visible to all waves
    if (ch + 1 < NCHUNK)
      issue(t0 + CHUNK, (ch & 1) ? TQ0 : TQ1, (ch & 1) ? TK0 : TK1, (ch & 1) ? TV0 : TV1);

    // gates: single barrier (Gc/lam/rdec on lane 0, beta on lanes 0..15)
    if (tid == 0) {
      float accg = 0.f;
      for (int t = 0; t < 16; ++t) {
        accg += gbuf[(size_t)(t0 + t) * HV + h];
        Gc[t]  = accg;
        lam[t] = __expf(accg);
      }
      for (int t = 0; t < 16; ++t) rdec[t] = __expf(Gc[15] - Gc[t]);
    }
    if (tid < 16) bet[tid] = betabuf[(size_t)(t0 + tid) * HV + h];
    __syncthreads();

    {  // lambda-scaled bf16 copies from raw tiles
      int r = tid >> 3, cs = (tid & 7) << 4;
      float lr = lam[r];
#pragma unroll
      for (int j = 0; j < 16; ++j) {
        Qs[r * 128 + cs + j] = f2b(b2f(tq[r * 128 + cs + j]) * lr);
        Ks[r * 128 + cs + j] = f2b(b2f(tk[r * 128 + cs + j]) * lr);
      }
    }
    __syncthreads();

    // --- WMMA stage A: Oin=(L*Q)@S0, RHS=(L*K)@S0, P=QK^T/KK^T ---
    v8f oacc[2], racc[2];
    oacc[0] = vzero8(); oacc[1] = vzero8();
    racc[0] = vzero8(); racc[1] = vzero8();
#pragma unroll
    for (int ks = 0; ks < 4; ++ks) {
      v16bf aQ = lda_frag(Qs + ks * 32, 128);
      v16bf aK = lda_frag(Ks + ks * 32, 128);
#pragma unroll
      for (int m = 0; m < 2; ++m) {
        int nt = wv + m * 4;
        v16bf bS = ldb_frag(Sb + nt * 16 * 128 + ks * 32, 128);
        oacc[m] = wmma_bf16(aQ, bS, oacc[m]);
        racc[m] = wmma_bf16(aK, bS, racc[m]);
      }
    }
    {
      int hi = (lane >> 4) << 3;
      int cl = lane & 15;
#pragma unroll
      for (int m = 0; m < 2; ++m) {
        int nt = wv + m * 4;
#pragma unroll
        for (int i = 0; i < 8; ++i) {
          Oin[(hi + i) * 128 + nt * 16 + cl] = oacc[m][i];
          RHS[(hi + i) * 128 + nt * 16 + cl] = racc[m][i];
        }
      }
    }
    if (wv < 2) {
      const u16* Ab = (wv == 0) ? tq : tk;
      v8f p = vzero8();
#pragma unroll
      for (int ks = 0; ks < 4; ++ks)
        p = wmma_bf16(lda_frag(Ab + ks * 32, 128), ldb_frag(tk + ks * 32, 128), p);
      float* Pd = (wv == 0) ? Pqk : Pkk;
      int hi = (lane >> 4) << 3;
      int cl = lane & 15;
#pragma unroll
      for (int i = 0; i < 8; ++i) Pd[(hi + i) * 16 + cl] = p[i];
    }
    __syncthreads();

    for (int e = tid; e < 256; e += 128) {
      int t = e >> 4, s = e & 15;
      float d = (s <= t) ? __expf(Gc[t] - Gc[s]) : 0.f;
      Msub[e] = (s < t) ? d * Pkk[e] : 0.f;
      Atl[t * 32 + s] = f2b((s <= t) ? d * Pqk[e] : 0.f);
    }
    __syncthreads();

    {  // forward substitution, column-parallel (no barriers)
      int v = tid;
      float Ucol[16];
#pragma unroll
      for (int t = 0; t < 16; ++t) {
        float a = b2f(tv[t * 128 + v]) - RHS[t * 128 + v];
        for (int s = 0; s < t; ++s) a -= Msub[t * 16 + s] * Ucol[s];
        Ucol[t] = bet[t] * a;
      }
#pragma unroll
      for (int t = 0; t < 16; ++t) Ubt[v * 32 + t] = f2b(Ucol[t]);
#pragma unroll
      for (int s = 0; s < 16; ++s) Kh[v * 32 + s] = f2b(rdec[s] * b2f(tk[s * 128 + v]));
    }
    __syncthreads();

    {  // O = Oin + tril@U -> global
      v16bf aT = lda_frag(Atl, 32);
      int hi = (lane >> 4) << 3;
      int cl = lane & 15;
#pragma unroll
      for (int m = 0; m < 2; ++m) {
        int nt = wv + m * 4;
        v8f c;
#pragma unroll
        for (int i = 0; i < 8; ++i) c[i] = Oin[(hi + i) * 128 + nt * 16 + cl];
        c = wmma_bf16(aT, ldb_frag(Ubt + nt * 16 * 32, 32), c);
#pragma unroll
        for (int i = 0; i < 8; ++i)
          o[(size_t)(t0 + hi + i) * VAL_DIM + h * DV + nt * 16 + cl] = c[i];
      }
    }
    {  // state: Sf = lamC*Sf + Kh @ U ; write f32 master AND bf16 copy for next chunk
      float lc = lam[15];
      int hi = (lane >> 4) << 3;
      int cl = lane & 15;
#pragma unroll
      for (int mi = 0; mi < 2; ++mi) {
        int m0 = (wv * 2 + mi) * 16;
        v16bf aKh = lda_frag(Kh + m0 * 32, 32);
#pragma unroll
        for (int nt = 0; nt < 8; ++nt) {
          int n0 = nt * 16;
          v8f c;
#pragma unroll
          for (int i = 0; i < 8; ++i) c[i] = lc * Sf[(n0 + cl) * 128 + m0 + hi + i];
          c = wmma_bf16(aKh, ldb_frag(Ubt + n0 * 32, 32), c);
#pragma unroll
          for (int i = 0; i < 8; ++i) {
            int sidx = (n0 + cl) * 128 + m0 + hi + i;
            Sf[sidx] = c[i];
            Sb[sidx] = f2b(c[i]);
          }
        }
      }
    }
    __syncthreads();
  }
}

// ---------------------------------------------------------------- gated RMSNorm * silu(z)
__global__ __launch_bounds__(128) void gatenorm_kernel(const float* __restrict__ o,
                                                       const float* __restrict__ qkvz,
                                                       const float* __restrict__ norm_w,
                                                       u16* __restrict__ og) {
  int t = blockIdx.x >> 5;
  int h = blockIdx.x & 31;
  int v = threadIdx.x;
  int lane = v & 31, wv = v >> 5;
  float x  = o[(size_t)t * VAL_DIM + h * 128 + v];
  float ss = x * x;
#pragma unroll
  for (int off = 16; off > 0; off >>= 1) ss += __shfl_xor(ss, off, 32);
  __shared__ float red[4];
  if (lane == 0) red[wv] = ss;
  __syncthreads();
  float var = (red[0] + red[1] + red[2] + red[3]) * (1.f / 128.f);
  float xs  = x * rsqrtf(var + 1e-6f) * norm_w[v];
  float z   = qkvz[(size_t)t * QKVZ_DIM + 8192 + h * 128 + v];
  og[(size_t)t * VAL_DIM + h * 128 + v] = f2b(xs * (z / (1.f + __expf(-z))));
}

// ---------------------------------------------------------------- host launch
extern "C" void kernel_launch(void* const* d_in, const int* in_sizes, int n_in,
                              void* d_out, int out_size, void* d_ws, size_t ws_size,
                              hipStream_t stream) {
  const float* X      = (const float*)d_in[0];
  const float* Wqkvz  = (const float*)d_in[1];
  const float* Wba    = (const float*)d_in[2];
  const float* convw  = (const float*)d_in[3];
  const float* A_log  = (const float*)d_in[4];
  const float* dt_b   = (const float*)d_in[5];
  const float* norm_w = (const float*)d_in[6];
  const float* Wout   = (const float*)d_in[7];
  float* out = (float*)d_out;
  (void)in_sizes; (void)n_in; (void)out_size; (void)ws_size;

  char* ws = (char*)d_ws;
  size_t off = 0;
  auto alloc = [&](size_t bytes) -> char* {
    char* p = ws + off;
    off += (bytes + 255) & ~(size_t)255;
    return p;
  };
  u16*   Xb    = (u16*)alloc((size_t)T_LEN * H_DIM * 2);
  u16*   WqbT  = (u16*)alloc((size_t)H_DIM * QKVZ_DIM * 2);   // [12288][2048]
  u16*   WbabT = (u16*)alloc((size_t)H_DIM * 64 * 2);         // [64][2048]
  u16*   WobT  = (u16*)alloc((size_t)VAL_DIM * H_DIM * 2);    // [2048][4096]
  float* qkvz  = (float*)alloc((size_t)T_LEN * QKVZ_DIM * 4);
  float* ba    = (float*)alloc((size_t)T_LEN * 64 * 4);
  float* qkf   = (float*)alloc((size_t)T_LEN * 4096 * 4);
  u16*   Qb    = (u16*)alloc((size_t)T_LEN * KEY_DIM * 2);
  u16*   Kb    = (u16*)alloc((size_t)T_LEN * KEY_DIM * 2);
  u16*   Vb    = (u16*)alloc((size_t)T_LEN * VAL_DIM * 2);
  float* gb    = (float*)alloc((size_t)T_LEN * HV * 4);
  float* beb   = (float*)alloc((size_t)T_LEN * HV * 4);
  float* ob    = (float*)alloc((size_t)T_LEN * VAL_DIM * 4);
  u16*   og    = (u16*)alloc((size_t)T_LEN * VAL_DIM * 2);

  cvt_bf16_kernel<<<dim3((T_LEN * H_DIM) / 256), dim3(256), 0, stream>>>(
      X, Xb, (long)T_LEN * H_DIM);
  cvt_t_kernel<<<dim3(QKVZ_DIM / 32, H_DIM / 32), dim3(256), 0, stream>>>(
      Wqkvz, WqbT, H_DIM, QKVZ_DIM);
  cvt_t_kernel<<<dim3(64 / 32, H_DIM / 32), dim3(256), 0, stream>>>(
      Wba, WbabT, H_DIM, 64);
  cvt_t_kernel<<<dim3(H_DIM / 32, VAL_DIM / 32), dim3(256), 0, stream>>>(
      Wout, WobT, VAL_DIM, H_DIM);

  gemm_bf16_kernel<<<dim3(QKVZ_DIM / 128, T_LEN / 128), dim3(256), 0, stream>>>(
      Xb, WqbT, qkvz, T_LEN, QKVZ_DIM, H_DIM);
  gemm_bf16_kernel<<<dim3(1, T_LEN / 128), dim3(256), 0, stream>>>(
      Xb, WbabT, ba, T_LEN, 64, H_DIM);

  conv_silu_kernel<<<dim3((T_LEN / 4) * CONV_DIM / 256, 1, 1), dim3(256), 0, stream>>>(
      qkvz, convw, qkf, Vb);
  qknorm_kernel<<<dim3(T_LEN, 32), dim3(128), 0, stream>>>(qkf, Qb, Kb);
  gba_kernel<<<dim3((T_LEN * HV + 255) / 256), dim3(256), 0, stream>>>(
      ba, A_log, dt_b, gb, beb);

  recurrence_kernel<<<dim3(HV), dim3(128), 168192, stream>>>(Qb, Kb, Vb, gb, beb, ob);

  gatenorm_kernel<<<dim3(T_LEN * HV), dim3(128), 0, stream>>>(ob, qkvz, norm_w, og);

  gemm_bf16_kernel<<<dim3(H_DIM / 128, T_LEN / 128), dim3(256), 0, stream>>>(
      og, WobT, out, T_LEN, H_DIM, VAL_DIM);
}
